// LDS_26328149524591
// MI455X (gfx1250) — matrix operand order repaired and checked by reading, not compiled, not run on hardware
//
#include <hip/hip_runtime.h>

// Problem constants (from reference)
#define STATE_DIM 1024
#define INPUT_DIM 512
#define OUTPUT_DIM 512
#define KX 10
#define BSZ 16
#define SEQ_LEN 2048
#define ROWS (BSZ * SEQ_LEN)   // 32768

typedef __attribute__((ext_vector_type(8)))  float  v8f;
typedef __attribute__((ext_vector_type(16))) __bf16 v16bf;

union FragBF {
    v16bf v;
    uint4 q[2];
};

// fp32 -> bf16 (round-to-nearest-even), raw u16 bits
__device__ __forceinline__ unsigned short f2bf(float f) {
    unsigned int u = __float_as_uint(f);
    unsigned int r = u + 0x7FFFu + ((u >> 16) & 1u);
    return (unsigned short)(r >> 16);
}

// ---------------------------------------------------------------------------
// Prep: fp32 -> bf16, 8 elements/thread, vectorized 16B in/out
// ---------------------------------------------------------------------------
__global__ __launch_bounds__(256) void cvt_bf16_kernel(const float* __restrict__ x,
                                                       unsigned short* __restrict__ y,
                                                       int n) {
    int i = (blockIdx.x * 256 + threadIdx.x) * 8;
    if (i >= n) return;
    float4 f0 = *(const float4*)&x[i];
    float4 f1 = *(const float4*)&x[i + 4];
    uint4 o;
    unsigned short* p = (unsigned short*)&o;
    p[0] = f2bf(f0.x); p[1] = f2bf(f0.y); p[2] = f2bf(f0.z); p[3] = f2bf(f0.w);
    p[4] = f2bf(f1.x); p[5] = f2bf(f1.y); p[6] = f2bf(f1.z); p[7] = f2bf(f1.w);
    *(uint4*)&y[i] = o;
}

// Bt[h][i] = B[i][h]
__global__ __launch_bounds__(256) void xpose_b_kernel(const float* __restrict__ B,
                                                      unsigned short* __restrict__ Bt) {
    int i = blockIdx.x * 256 + threadIdx.x;    // over 1024*512
    if (i >= STATE_DIM * INPUT_DIM) return;
    int ii = i & (INPUT_DIM - 1);
    int h  = i >> 9;
    Bt[i] = f2bf(B[(size_t)ii * STATE_DIM + h]);
}

// Ct[o][h] = C[h][o]
__global__ __launch_bounds__(256) void xpose_c_kernel(const float* __restrict__ C,
                                                      unsigned short* __restrict__ Ct) {
    int i = blockIdx.x * 256 + threadIdx.x;    // over 512*1024
    if (i >= OUTPUT_DIM * STATE_DIM) return;
    int h = i & (STATE_DIM - 1);
    int o = i >> 10;
    Ct[i] = f2bf(C[(size_t)h * OUTPUT_DIM + o]);
}

// Mt[j][o][i] = M[o][i][j]
__global__ __launch_bounds__(256) void xpose_m_kernel(const float* __restrict__ M,
                                                      unsigned short* __restrict__ Mt) {
    int i = blockIdx.x * 256 + threadIdx.x;    // over 10*512*512
    if (i >= KX * OUTPUT_DIM * INPUT_DIM) return;
    int ii = i & (INPUT_DIM - 1);
    int o  = (i >> 9) & (OUTPUT_DIM - 1);
    int j  = i >> 18;
    Mt[i] = f2bf(M[((size_t)o * INPUT_DIM + ii) * KX + j]);
}

// ---------------------------------------------------------------------------
// Shared WMMA micro-kernel: one K=32 step, wave computes a 32x32 tile
// As: [128 rows][40], Bs: [64 n][40]; both bf16, padded stride 40 (bank-safe)
// ---------------------------------------------------------------------------
#define APAD 40
#define A_TILE (128 * APAD)
#define B_TILE (64 * APAD)

__device__ __forceinline__ void wmma_step(const unsigned short* __restrict__ As,
                                          const unsigned short* __restrict__ Bs,
                                          int wr, int wc, int lane, v8f acc[2][2]) {
    const int mrow  = lane & 15;
    const int khalf = (lane >> 4) * 8;    // A: K-half select per ISA layout
    const int bk    = (lane >> 4) * 8;    // B: K-half select per ISA layout
    FragBF a0, a1, b0, b1;
    const unsigned short* ar0 = &As[(wr * 32 + mrow) * APAD];
    const unsigned short* ar1 = ar0 + 16 * APAD;
    a0.q[0] = *(const uint4*)(ar0 + khalf);
    a0.q[1] = *(const uint4*)(ar0 + 16 + khalf);
    a1.q[0] = *(const uint4*)(ar1 + khalf);
    a1.q[1] = *(const uint4*)(ar1 + 16 + khalf);
    const unsigned short* br0 = &Bs[(wc * 32 + mrow) * APAD + bk];
    const unsigned short* br1 = br0 + 16 * APAD;
    b0.q[0] = *(const uint4*)(br0);
    b0.q[1] = *(const uint4*)(br0 + 16);
    b1.q[0] = *(const uint4*)(br1);
    b1.q[1] = *(const uint4*)(br1 + 16);
    acc[0][0] = __builtin_amdgcn_wmma_f32_16x16x32_bf16(false, a0.v, false, b0.v, (short)0, acc[0][0], false, false);
    acc[0][1] = __builtin_amdgcn_wmma_f32_16x16x32_bf16(false, a0.v, false, b1.v, (short)0, acc[0][1], false, false);
    acc[1][0] = __builtin_amdgcn_wmma_f32_16x16x32_bf16(false, a1.v, false, b0.v, (short)0, acc[1][0], false, false);
    acc[1][1] = __builtin_amdgcn_wmma_f32_16x16x32_bf16(false, a1.v, false, b1.v, (short)0, acc[1][1], false, false);
}

__device__ __forceinline__ void store_tiles(unsigned short* __restrict__ As,
                                            unsigned short* __restrict__ Bs,
                                            int tid, const uint4 ra[2], uint4 rb) {
#pragma unroll
    for (int i = 0; i < 2; ++i) {
        int q = tid + i * 256;
        int r = q >> 2, c4 = q & 3;
        *(uint4*)&As[r * APAD + c4 * 8] = ra[i];
    }
    {
        int n = tid >> 2, c4 = tid & 3;
        *(uint4*)&Bs[n * APAD + c4 * 8] = rb;
    }
}

__device__ __forceinline__ void store_acc(float* __restrict__ out, int ldn,
                                          int row0, int col0, int wr, int wc, int lane,
                                          const v8f acc[2][2]) {
    const int orow = (lane >> 4) * 8, ocol = lane & 15;
#pragma unroll
    for (int ar = 0; ar < 2; ++ar)
#pragma unroll
        for (int bc = 0; bc < 2; ++bc) {
            float* op = out + (size_t)(row0 + wr * 32 + ar * 16) * ldn
                            + col0 + wc * 32 + bc * 16 + ocol;
#pragma unroll
            for (int v = 0; v < 8; ++v) op[(size_t)(orow + v) * ldn] = acc[ar][bc][v];
        }
}

// ---------------------------------------------------------------------------
// GEMM 1: uB[r][h] = sum_i Xb[r][i] * B[i][h]   (M=32768, K=512, N=1024)
// Block tile 128x64, 8 waves x (32x32), LDS double-buffered, reg pipeline.
// ---------------------------------------------------------------------------
__global__ __launch_bounds__(256) void gemm_uB_kernel(const unsigned short* __restrict__ Xb,
                                                      const unsigned short* __restrict__ Bt,
                                                      float* __restrict__ uB) {
    __shared__ alignas(16) unsigned short As[2][A_TILE];
    __shared__ alignas(16) unsigned short Bs[2][B_TILE];

    const int tid  = threadIdx.x;
    const int lane = tid & 31;
    const int wave = tid >> 5;
    const int wr   = wave >> 1;   // 0..3
    const int wc   = wave & 1;    // 0..1
    const int row0 = blockIdx.y * 128;
    const int col0 = blockIdx.x * 64;

    v8f acc[2][2] = {};

    auto load_a = [&](int k0, uint4 ra[2]) {
#pragma unroll
        for (int i = 0; i < 2; ++i) {
            int q = tid + i * 256;
            int r = q >> 2, c4 = q & 3;
            ra[i] = *(const uint4*)&Xb[(size_t)(row0 + r) * INPUT_DIM + k0 + c4 * 8];
        }
    };
    auto load_b = [&](int k0, uint4& rb) {
        int n = tid >> 2, c4 = tid & 3;
        rb = *(const uint4*)&Bt[(size_t)(col0 + n) * INPUT_DIM + k0 + c4 * 8];
    };

    uint4 ra[2], rb;
    load_a(0, ra);
    load_b(0, rb);
    int buf = 0;
    store_tiles(As[0], Bs[0], tid, ra, rb);
    __syncthreads();

    const int NSTEPS = INPUT_DIM / 32;   // 16
    for (int s = 1; s < NSTEPS; ++s) {
        uint4 na[2], nb;
        load_a(s * 32, na);
        load_b(s * 32, nb);
        wmma_step(As[buf], Bs[buf], wr, wc, lane, acc);
        store_tiles(As[buf ^ 1], Bs[buf ^ 1], tid, na, nb);
        __syncthreads();
        buf ^= 1;
    }
    wmma_step(As[buf], Bs[buf], wr, wc, lane, acc);

    store_acc(uB, STATE_DIM, row0, col0, wr, wc, lane, acc);
}

// ---------------------------------------------------------------------------
// Sequential scan: h_t = A*h_{t-1} + uB_t; thread <-> (b, h), coalesced over h.
// 8-deep load batching to amortize HBM latency around the serial fma chain.
// ---------------------------------------------------------------------------
__global__ __launch_bounds__(256) void scan_kernel(const float* __restrict__ uB,
                                                   const float* __restrict__ h0,
                                                   const float* __restrict__ A,
                                                   unsigned short* __restrict__ Hs) {
    const int h = blockIdx.x * 256 + threadIdx.x;   // 0..1023
    const int b = blockIdx.y;                        // 0..15
    const float a  = A[h];
    float       hc = h0[h];
    size_t base = ((size_t)b * SEQ_LEN) * STATE_DIM + h;
    for (int t = 0; t < SEQ_LEN; t += 8) {
        float u[8];
#pragma unroll
        for (int i = 0; i < 8; ++i) u[i] = uB[base + (size_t)i * STATE_DIM];
#pragma unroll
        for (int i = 0; i < 8; ++i) {
            hc = fmaf(a, hc, u[i]);
            Hs[base + (size_t)i * STATE_DIM] = f2bf(hc);
        }
        base += (size_t)8 * STATE_DIM;
    }
}

// ---------------------------------------------------------------------------
// GEMM 2 (fused): out = Hs @ C + causal-conv(inputs, M)
// K-steps: 32 for the C-GEMM (K=1024) + 10 lags x 16 steps (K=512 each) = 192
// ---------------------------------------------------------------------------
__global__ __launch_bounds__(256) void gemm_out_kernel(const unsigned short* __restrict__ Hs,
                                                       const unsigned short* __restrict__ Ct,
                                                       const unsigned short* __restrict__ Xb,
                                                       const unsigned short* __restrict__ Mt,
                                                       float* __restrict__ out) {
    __shared__ alignas(16) unsigned short As[2][A_TILE];
    __shared__ alignas(16) unsigned short Bs[2][B_TILE];

    const int tid  = threadIdx.x;
    const int lane = tid & 31;
    const int wave = tid >> 5;
    const int wr   = wave >> 1;
    const int wc   = wave & 1;
    const int row0 = blockIdx.y * 128;
    const int col0 = blockIdx.x * 64;

    v8f acc[2][2] = {};

    auto load_a = [&](int s, uint4 ra[2]) {
        if (s < 32) {                       // C-GEMM part: A = Hs, K=1024
            int k0 = s * 32;
#pragma unroll
            for (int i = 0; i < 2; ++i) {
                int q = tid + i * 256;
                int r = q >> 2, c4 = q & 3;
                ra[i] = *(const uint4*)&Hs[(size_t)(row0 + r) * STATE_DIM + k0 + c4 * 8];
            }
        } else {                            // AR part: A = Xb shifted by lag j
            int s2 = s - 32;
            int j  = s2 >> 4;
            int k0 = (s2 & 15) * 32;
#pragma unroll
            for (int i = 0; i < 2; ++i) {
                int q = tid + i * 256;
                int r = q >> 2, c4 = q & 3;
                int rg = row0 + r;
                int t  = rg & (SEQ_LEN - 1);
                uint4 v = {0u, 0u, 0u, 0u};
                if (t >= j)
                    v = *(const uint4*)&Xb[(size_t)(rg - j) * INPUT_DIM + k0 + c4 * 8];
                ra[i] = v;
            }
        }
    };
    auto load_b = [&](int s, uint4& rb) {
        int n = tid >> 2, c4 = tid & 3;
        if (s < 32) {
            int k0 = s * 32;
            rb = *(const uint4*)&Ct[(size_t)(col0 + n) * STATE_DIM + k0 + c4 * 8];
        } else {
            int s2 = s - 32;
            int j  = s2 >> 4;
            int k0 = (s2 & 15) * 32;
            rb = *(const uint4*)&Mt[(size_t)j * OUTPUT_DIM * INPUT_DIM
                                    + (size_t)(col0 + n) * INPUT_DIM + k0 + c4 * 8];
        }
    };

    uint4 ra[2], rb;
    load_a(0, ra);
    load_b(0, rb);
    int buf = 0;
    store_tiles(As[0], Bs[0], tid, ra, rb);
    __syncthreads();

    const int NSTEPS = 32 + KX * (INPUT_DIM / 32);   // 192
    for (int s = 1; s < NSTEPS; ++s) {
        uint4 na[2], nb;
        load_a(s, na);
        load_b(s, nb);
        wmma_step(As[buf], Bs[buf], wr, wc, lane, acc);
        store_tiles(As[buf ^ 1], Bs[buf ^ 1], tid, na, nb);
        __syncthreads();
        buf ^= 1;
    }
    wmma_step(As[buf], Bs[buf], wr, wc, lane, acc);

    store_acc(out, OUTPUT_DIM, row0, col0, wr, wc, lane, acc);
}

// ---------------------------------------------------------------------------
// Host launcher
// ---------------------------------------------------------------------------
extern "C" void kernel_launch(void* const* d_in, const int* in_sizes, int n_in,
                              void* d_out, int out_size, void* d_ws, size_t ws_size,
                              hipStream_t stream) {
    const float* inputs = (const float*)d_in[0];  // (16, 2048, 512)
    const float* h0     = (const float*)d_in[1];  // (1024,)
    const float* A      = (const float*)d_in[2];  // (1024,)
    const float* B      = (const float*)d_in[3];  // (512, 1024)
    const float* C      = (const float*)d_in[4];  // (1024, 512)
    const float* M      = (const float*)d_in[5];  // (512, 512, 10)
    float*       out    = (float*)d_out;          // (16, 2048, 512)

    char* ws = (char*)d_ws;
    unsigned short* Xb = (unsigned short*)ws; ws += (size_t)ROWS * INPUT_DIM * 2;            // 33.5 MB
    unsigned short* Bt = (unsigned short*)ws; ws += (size_t)STATE_DIM * INPUT_DIM * 2;       // 1 MB
    unsigned short* Ct = (unsigned short*)ws; ws += (size_t)OUTPUT_DIM * STATE_DIM * 2;      // 1 MB
    unsigned short* Mt = (unsigned short*)ws; ws += (size_t)KX * OUTPUT_DIM * INPUT_DIM * 2; // 5 MB
    float*          uB = (float*)ws;          ws += (size_t)ROWS * STATE_DIM * 4;            // 134 MB
    unsigned short* Hs = (unsigned short*)ws;                                                // 67 MB

    // Prep: bf16 cast of inputs + transposed bf16 weights
    {
        int n = ROWS * INPUT_DIM;
        cvt_bf16_kernel<<<(n / 8 + 255) / 256, 256, 0, stream>>>(inputs, Xb, n);
    }
    {
        int n = STATE_DIM * INPUT_DIM;
        xpose_b_kernel<<<(n + 255) / 256, 256, 0, stream>>>(B, Bt);
    }
    {
        int n = OUTPUT_DIM * STATE_DIM;
        xpose_c_kernel<<<(n + 255) / 256, 256, 0, stream>>>(C, Ct);
    }
    {
        int n = KX * OUTPUT_DIM * INPUT_DIM;
        xpose_m_kernel<<<(n + 255) / 256, 256, 0, stream>>>(M, Mt);
    }

    // uB = inputs @ B (WMMA bf16)
    gemm_uB_kernel<<<dim3(STATE_DIM / 64, ROWS / 128), 256, 0, stream>>>(Xb, Bt, uB);

    // sequential diagonal-state scan
    scan_kernel<<<dim3(STATE_DIM / 256, BSZ), 256, 0, stream>>>(uB, h0, A, Hs);

    // out = Hs @ C + AR(inputs, M)  (fused WMMA bf16)
    gemm_out_kernel<<<dim3(OUTPUT_DIM / 64, ROWS / 128), 256, 0, stream>>>(Hs, Ct, Xb, Mt, out);
}
